// SlowLayer_695784702460
// MI455X (gfx1250) — matrix operand order repaired and checked by reading, not compiled, not run on hardware
//
#include <hip/hip_runtime.h>
#include <hip/hip_bf16.h>

#define DM   1024
#define DS   64
#define FH   2048
#define NE   8
#define NB   4
#define TT   2048
#define NTOK 8192
#define NCH  32
#define CAP  17408   // 16384 assignments + per-expert pad to 128 (136 * 128)

typedef __bf16 bh;
typedef __attribute__((ext_vector_type(16))) bh    v16bf;
typedef __attribute__((ext_vector_type(8)))  float v8f;

union Frag { v16bf v; uint4 q[2]; };

__device__ inline unsigned short f2bf(float f) {
  union { float f; unsigned u; } a; a.f = f;
  unsigned r = a.u + 0x7FFFu + ((a.u >> 16) & 1u);
  return (unsigned short)(r >> 16);
}
__device__ inline float bf2f(unsigned short s) {
  union { unsigned u; float f; } a; a.u = ((unsigned)s) << 16; return a.f;
}
// A fragment (16x32 bf16): lane<16 holds K {0..15}, lane>=16 holds {16..31}
__device__ inline v16bf load_a(const unsigned short* p) {
  Frag f; f.q[0] = *(const uint4*)p; f.q[1] = *(const uint4*)(p + 8); return f.v;
}
// B fragment (32x16 bf16, from N-major storage): lane half holds 16 contiguous K
__device__ inline v16bf load_b(const unsigned short* p) {
  Frag f; f.q[0] = *(const uint4*)p; f.q[1] = *(const uint4*)(p + 8); return f.v;
}
__device__ inline v16bf zfrag() {
  Frag f; f.q[0] = make_uint4(0,0,0,0); f.q[1] = make_uint4(0,0,0,0); return f.v;
}
__device__ inline v8f wmma_bf16(v16bf a, v16bf b, v8f c) {
  return __builtin_amdgcn_wmma_f32_16x16x32_bf16(false, a, false, b, (short)0, c,
                                                 false, false);
}

// ---------------------------------------------------------------- transpose+cvt
__global__ __launch_bounds__(256) void transpose_bf16_kernel(
    const float* __restrict__ in, unsigned short* __restrict__ out,
    int R, int C, size_t ibs, size_t obs) {
  __shared__ float tile[32][33];
  const float* inp = in + (size_t)blockIdx.z * ibs;
  unsigned short* outp = out + (size_t)blockIdx.z * obs;
  int tx = threadIdx.x & 31, ty = threadIdx.x >> 5;
  int c0 = blockIdx.x * 32, r0 = blockIdx.y * 32;
#pragma unroll
  for (int i = 0; i < 32; i += 8) {
    int r = r0 + ty + i, c = c0 + tx;
    tile[ty + i][tx] = (r < R && c < C) ? inp[(size_t)r * C + c] : 0.f;
  }
  __syncthreads();
#pragma unroll
  for (int i = 0; i < 32; i += 8) {
    int c = c0 + ty + i, r = r0 + tx;
    if (c < C && r < R) outp[(size_t)c * R + r] = f2bf(tile[tx][ty + i]);
  }
}

__global__ void init_counts_kernel(int* counts) {
  if (threadIdx.x < NE) counts[threadIdx.x] = 0;
}

// ---------------------------------------------------------------- compnorm
__device__ inline float blk_reduce(float v, float* red, bool domax) {
  int tid = threadIdx.x;
  red[tid] = v; __syncthreads();
  for (int s = 128; s > 0; s >>= 1) {
    if (tid < s) red[tid] = domax ? fmaxf(red[tid], red[tid + s]) : (red[tid] + red[tid + s]);
    __syncthreads();
  }
  float r = red[0]; __syncthreads(); return r;
}

__global__ __launch_bounds__(256) void compnorm_kernel(
    const float* __restrict__ x, const float* __restrict__ tau,
    const float* __restrict__ scale, unsigned short* __restrict__ out) {
  __shared__ float red[256];
  const int t = blockIdx.x, tid = threadIdx.x;
  const float4 v = ((const float4*)(x + (size_t)t * DM))[tid];
  float s  = v.x + v.y + v.z + v.w;
  float sq = v.x * v.x + v.y * v.y + v.z * v.z + v.w * v.w;
  float mx = fmaxf(fmaxf(v.x, v.y), fmaxf(v.z, v.w));
  s  = blk_reduce(s,  red, false);
  sq = blk_reduce(sq, red, false);
  mx = blk_reduce(mx, red, true);
  const float mean = s * (1.f / DM);
  const float rms  = rsqrtf(sq * (1.f / DM) + 1e-8f);
  const float tp   = fmaxf(tau[0], 1.f);
  const float mc   = mx - mean;           // max of centered row
  float4 e;
  e.x = expf((v.x - mean - mc) / tp);
  e.y = expf((v.y - mean - mc) / tp);
  e.z = expf((v.z - mean - mc) / tp);
  e.w = expf((v.w - mean - mc) / tp);
  const float Z = blk_reduce(e.x + e.y + e.z + e.w, red, false);
  const float4 sc = ((const float4*)scale)[tid];
  const float k = rms * (float)DM / Z;
  ushort4 o;
  o.x = f2bf(v.x * e.x * sc.x * k);
  o.y = f2bf(v.y * e.y * sc.y * k);
  o.z = f2bf(v.z * e.z * sc.z * k);
  o.w = f2bf(v.w * e.w * sc.w * k);
  ((ushort4*)(out + (size_t)t * DM))[tid] = o;
}

// ---------------------------------------------------------------- proj GEMM + gate
// u = sigmoid(h1@Win[:,64:] + b) * tanh(h1@Win[:,:64] + b)
// K pipelined with ping-pong register buffers (no rotation copies)
__global__ __launch_bounds__(256) void gemm_proj_kernel(
    const unsigned short* __restrict__ A,   // h1 [NTOK][DM] bf16
    const unsigned short* __restrict__ Bt,  // WinT [128][DM] bf16 (N-major)
    const float* __restrict__ b_in,         // [128]
    float* __restrict__ u) {                // [NTOK][DS]
  const int tid = threadIdx.x, wid = tid >> 5, lane = tid & 31;
  const int m0 = blockIdx.x * 128 + wid * 16;
  const int l15 = lane & 15, hi = lane >> 4;
  const unsigned short* ap = A + (size_t)(m0 + l15) * DM + hi * 16;
  const unsigned short* bp[8];
#pragma unroll
  for (int j = 0; j < 8; ++j) bp[j] = Bt + (size_t)(j * 16 + l15) * DM + hi * 16;
  v8f acc[8];
#pragma unroll
  for (int j = 0; j < 8; ++j)
#pragma unroll
    for (int r = 0; r < 8; ++r) acc[j][r] = 0.f;

  v16bf a0, a1, b0[8], b1[8];
  a0 = load_a(ap);
#pragma unroll
  for (int j = 0; j < 8; ++j) b0[j] = load_b(bp[j]);
  for (int kk = 0; kk < DM - 64; kk += 64) {
    a1 = load_a(ap + kk + 32);
#pragma unroll
    for (int j = 0; j < 8; ++j) b1[j] = load_b(bp[j] + kk + 32);
#pragma unroll
    for (int j = 0; j < 8; ++j) acc[j] = wmma_bf16(a0, b0[j], acc[j]);
    a0 = load_a(ap + kk + 64);
#pragma unroll
    for (int j = 0; j < 8; ++j) b0[j] = load_b(bp[j] + kk + 64);
#pragma unroll
    for (int j = 0; j < 8; ++j) acc[j] = wmma_bf16(a1, b1[j], acc[j]);
  }
  a1 = load_a(ap + DM - 32);
#pragma unroll
  for (int j = 0; j < 8; ++j) b1[j] = load_b(bp[j] + DM - 32);
#pragma unroll
  for (int j = 0; j < 8; ++j) acc[j] = wmma_bf16(a0, b0[j], acc[j]);
#pragma unroll
  for (int j = 0; j < 8; ++j) acc[j] = wmma_bf16(a1, b1[j], acc[j]);

  const int mb = m0 + (hi ? 8 : 0);
#pragma unroll
  for (int j = 0; j < 4; ++j) {
    const int d = j * 16 + l15;
    const float bi = b_in[d], bg = b_in[DS + d];
#pragma unroll
    for (int r = 0; r < 8; ++r) {
      float iv = tanhf(acc[j][r] + bi);
      float gv = 1.f / (1.f + expf(-(acc[j + 4][r] + bg)));
      u[(size_t)(mb + r) * DS + d] = gv * iv;
    }
  }
}

// ---------------------------------------------------------------- LRU scan
__global__ __launch_bounds__(256) void lru_chunk_kernel(
    const float* __restrict__ u, const float* __restrict__ rec_w,
    const float* __restrict__ pos_bias,
    float* __restrict__ states, float* __restrict__ chunk_last) {
  const int idx = blockIdx.x * 256 + threadIdx.x;   // NB*NCH*DS = 8192
  const int d = idx & 63, c = (idx >> 6) & 31, b = idx >> 11;
  const float a = 1.f / (1.f + expf(-(rec_w[d] + pos_bias[d])));
  const float* up = u + ((size_t)b * TT + c * 64) * DS + d;
  float* sp = states + ((size_t)b * TT + c * 64) * DS + d;
  float h = 0.f;
#pragma unroll 4
  for (int j = 0; j < 64; ++j) { h = a * h + up[(size_t)j * DS]; sp[(size_t)j * DS] = h; }
  chunk_last[(b * NCH + c) * DS + d] = h;
}

__global__ __launch_bounds__(256) void lru_combine_kernel(
    const float* __restrict__ rec_w, const float* __restrict__ pos_bias,
    const float* __restrict__ state0, const float* __restrict__ chunk_last,
    float* __restrict__ chunk_start, float* __restrict__ out_state) {
  const int idx = threadIdx.x;       // 256 = NB*DS
  const int d = idx & 63, b = idx >> 6;
  const float a = 1.f / (1.f + expf(-(rec_w[d] + pos_bias[d])));
  const float a64 = powf(a, 64.f);
  float carry = state0[b * DS + d];
  for (int c = 0; c < NCH; ++c) {
    chunk_start[(b * NCH + c) * DS + d] = carry;
    carry = carry * a64 + chunk_last[(b * NCH + c) * DS + d];
  }
  out_state[b * DS + d] = carry;
}

__global__ __launch_bounds__(256) void lru_fixup_kernel(
    const float* __restrict__ states, const float* __restrict__ chunk_start,
    const float* __restrict__ rec_w, const float* __restrict__ pos_bias,
    unsigned short* __restrict__ statesbf) {
  const int idx = blockIdx.x * 256 + threadIdx.x;   // NB*TT*DS = 524288
  const int d = idx & 63, t = (idx >> 6) & (TT - 1), b = idx >> 17;
  const int c = t >> 6, j = t & 63;
  const float a = 1.f / (1.f + expf(-(rec_w[d] + pos_bias[d])));
  float s = states[idx] + chunk_start[(b * NCH + c) * DS + d] * powf(a, (float)(j + 1));
  statesbf[idx] = f2bf(s);
}

// ---------------------------------------------------------------- out GEMM + residual
// K = 64: fully unrolled, all fragments loaded before any WMMA
__global__ __launch_bounds__(256) void gemm_out_kernel(
    const unsigned short* __restrict__ A,    // statesbf [NTOK][DS]
    const unsigned short* __restrict__ Bt,   // WoutT [DM][DS]
    const float* __restrict__ b_out, const float* __restrict__ x,
    float* __restrict__ xout) {
  const int tid = threadIdx.x, wid = tid >> 5, lane = tid & 31;
  const int m0 = blockIdx.x * 128 + wid * 16;
  const int n0 = blockIdx.y * 128;
  const int l15 = lane & 15, hi = lane >> 4;
  const unsigned short* ap = A + (size_t)(m0 + l15) * DS + hi * 16;
  const unsigned short* bp[8];
#pragma unroll
  for (int j = 0; j < 8; ++j) bp[j] = Bt + (size_t)(n0 + j * 16 + l15) * DS + hi * 16;
  v8f acc[8];
#pragma unroll
  for (int j = 0; j < 8; ++j)
#pragma unroll
    for (int r = 0; r < 8; ++r) acc[j][r] = 0.f;
  v16bf a0 = load_a(ap), a1 = load_a(ap + 32);
  v16bf b0[8], b1[8];
#pragma unroll
  for (int j = 0; j < 8; ++j) { b0[j] = load_b(bp[j]); b1[j] = load_b(bp[j] + 32); }
#pragma unroll
  for (int j = 0; j < 8; ++j) acc[j] = wmma_bf16(a0, b0[j], acc[j]);
#pragma unroll
  for (int j = 0; j < 8; ++j) acc[j] = wmma_bf16(a1, b1[j], acc[j]);
  const int mb = m0 + (hi ? 8 : 0);
#pragma unroll
  for (int j = 0; j < 8; ++j) {
    const int n = n0 + j * 16 + l15;
    const float bo = b_out[n];
#pragma unroll
    for (int r = 0; r < 8; ++r) {
      const int m = mb + r;
      xout[(size_t)m * DM + n] = x[(size_t)m * DM + n] + acc[j][r] + bo;
    }
  }
}

// ---------------------------------------------------------------- router
__global__ __launch_bounds__(256) void router_kernel(
    const unsigned short* __restrict__ h2, const float* __restrict__ Wr, // [DM][NE]
    int* __restrict__ counts, float* __restrict__ Ppart,                  // [1024][NE]
    int* __restrict__ tok_e, float* __restrict__ tok_w) {
  __shared__ float sP[NE];
  const int tid = threadIdx.x, wid = tid >> 5, lane = tid & 31;
  if (tid < NE) sP[tid] = 0.f;
  __syncthreads();
  const int t = blockIdx.x * 8 + wid;
  const unsigned short* hp = h2 + (size_t)t * DM;
  float acc[NE];
#pragma unroll
  for (int e = 0; e < NE; ++e) acc[e] = 0.f;
  for (int k = lane; k < DM; k += 32) {
    const float hv = bf2f(hp[k]);
    const float* wr = Wr + k * NE;
#pragma unroll
    for (int e = 0; e < NE; ++e) acc[e] += hv * wr[e];
  }
#pragma unroll
  for (int off = 16; off > 0; off >>= 1)
#pragma unroll
    for (int e = 0; e < NE; ++e) acc[e] += __shfl_xor(acc[e], off, 32);
  if (lane == 0) {
    float mx = acc[0];
#pragma unroll
    for (int e = 1; e < NE; ++e) mx = fmaxf(mx, acc[e]);
    float p[NE], Z = 0.f;
#pragma unroll
    for (int e = 0; e < NE; ++e) { p[e] = expf(acc[e] - mx); Z += p[e]; }
#pragma unroll
    for (int e = 0; e < NE; ++e) p[e] /= Z;
    int i0 = 0;
#pragma unroll
    for (int e = 1; e < NE; ++e) if (p[e] > p[i0]) i0 = e;
    int i1 = (i0 == 0) ? 1 : 0;
#pragma unroll
    for (int e = 0; e < NE; ++e) if (e != i0 && p[e] > p[i1]) i1 = e;
    const float sw = p[i0] + p[i1];
    tok_e[2 * t] = i0; tok_e[2 * t + 1] = i1;
    tok_w[2 * t] = p[i0] / sw; tok_w[2 * t + 1] = p[i1] / sw;
    atomicAdd(&counts[i0], 1); atomicAdd(&counts[i1], 1);
#pragma unroll
    for (int e = 0; e < NE; ++e) atomicAdd(&sP[e], p[e]);
  }
  __syncthreads();
  if (tid < NE) Ppart[blockIdx.x * NE + tid] = sP[tid];
}

__global__ __launch_bounds__(256) void router_setup_kernel(
    const int* __restrict__ counts, const float* __restrict__ Ppart,
    int* __restrict__ base_pad, int* __restrict__ counts2,
    int* __restrict__ assign_tok, float* __restrict__ aux_out) {
  __shared__ float Ps[NE];
  const int tid = threadIdx.x;
  if (tid < NE) {
    float s = 0.f;
    for (int b = 0; b < 1024; ++b) s += Ppart[b * NE + tid];   // fixed-order, deterministic
    Ps[tid] = s;
    counts2[tid] = 0;
  }
  __syncthreads();
  if (tid == 0) {
    int off = 0;
    // segments padded to 128 rows (block M-tile of the MoE GEMMs)
    for (int e = 0; e < NE; ++e) { base_pad[e] = off; off += ((counts[e] + 127) >> 7) << 7; }
    base_pad[NE] = off;
    float aux = 0.f;
    for (int e = 0; e < NE; ++e)
      aux += ((float)counts[e] / (float)(NTOK * 2)) * (Ps[e] / (float)NTOK);
    aux_out[0] = (float)NE * aux;
  }
  for (int i = tid; i < CAP; i += 256) assign_tok[i] = -1;
}

__global__ __launch_bounds__(256) void router_scatter_kernel(
    const int* __restrict__ tok_e, int* __restrict__ counts2,
    const int* __restrict__ base_pad, int* __restrict__ assign_tok,
    int* __restrict__ slot_of) {
  const int t = blockIdx.x * 256 + threadIdx.x;   // NTOK
#pragma unroll
  for (int j = 0; j < 2; ++j) {
    const int e = tok_e[2 * t + j];
    const int pos = base_pad[e] + atomicAdd(&counts2[e], 1);
    assign_tok[pos] = t;
    slot_of[2 * t + j] = pos;
  }
}

// ---------------------------------------------------------------- MoE GEMM1 (silu)
// wave tile 64M x 64N (16 WMMA per 8 fragment loads), block 128M x 256N
__global__ __launch_bounds__(256) void gemm1_moe_kernel(
    const unsigned short* __restrict__ h2,    // [NTOK][DM]
    const unsigned short* __restrict__ We1T,  // [NE][FH][DM]
    const float* __restrict__ be1,            // [NE][FH]
    const int* __restrict__ assign_tok, const int* __restrict__ base_pad,
    unsigned short* __restrict__ Hbuf) {      // [CAP][FH]
  const int row0 = blockIdx.x * 128;
  if (row0 >= base_pad[NE]) return;
  int e = 0;
#pragma unroll
  for (int i = 0; i < NE; ++i) if (row0 >= base_pad[i + 1]) e = i + 1;
  const int tid = threadIdx.x, wid = tid >> 5, lane = tid & 31;
  const int wm = wid >> 2, wn = wid & 3;
  const int l15 = lane & 15, hi = lane >> 4;
  const unsigned short* ap[4];
#pragma unroll
  for (int i = 0; i < 4; ++i) {
    const int r = row0 + wm * 64 + i * 16 + l15;
    const int tok = assign_tok[r];
    ap[i] = (tok >= 0) ? (h2 + (size_t)tok * DM + hi * 16) : nullptr;
  }
  const int n0 = blockIdx.y * 256 + wn * 64;
  const unsigned short* bbase = We1T + (size_t)e * FH * DM;
  const unsigned short* bp[4];
#pragma unroll
  for (int j = 0; j < 4; ++j) bp[j] = bbase + (size_t)(n0 + j * 16 + l15) * DM + hi * 16;
  v8f acc[4][4];
#pragma unroll
  for (int i = 0; i < 4; ++i)
#pragma unroll
    for (int j = 0; j < 4; ++j)
#pragma unroll
      for (int r = 0; r < 8; ++r) acc[i][j][r] = 0.f;

  v16bf a0[4], a1[4], b0[4], b1[4];
#pragma unroll
  for (int i = 0; i < 4; ++i) a0[i] = ap[i] ? load_a(ap[i]) : zfrag();
#pragma unroll
  for (int j = 0; j < 4; ++j) b0[j] = load_b(bp[j]);
  for (int kk = 0; kk < DM - 64; kk += 64) {
#pragma unroll
    for (int i = 0; i < 4; ++i) a1[i] = ap[i] ? load_a(ap[i] + kk + 32) : zfrag();
#pragma unroll
    for (int j = 0; j < 4; ++j) b1[j] = load_b(bp[j] + kk + 32);
#pragma unroll
    for (int j = 0; j < 4; ++j) __builtin_prefetch(bp[j] + kk + 128, 0, 1);
#pragma unroll
    for (int i = 0; i < 4; ++i)
#pragma unroll
      for (int j = 0; j < 4; ++j) acc[i][j] = wmma_bf16(a0[i], b0[j], acc[i][j]);
#pragma unroll
    for (int i = 0; i < 4; ++i) a0[i] = ap[i] ? load_a(ap[i] + kk + 64) : zfrag();
#pragma unroll
    for (int j = 0; j < 4; ++j) b0[j] = load_b(bp[j] + kk + 64);
#pragma unroll
    for (int i = 0; i < 4; ++i)
#pragma unroll
      for (int j = 0; j < 4; ++j) acc[i][j] = wmma_bf16(a1[i], b1[j], acc[i][j]);
  }
#pragma unroll
  for (int i = 0; i < 4; ++i) a1[i] = ap[i] ? load_a(ap[i] + DM - 32) : zfrag();
#pragma unroll
  for (int j = 0; j < 4; ++j) b1[j] = load_b(bp[j] + DM - 32);
#pragma unroll
  for (int i = 0; i < 4; ++i)
#pragma unroll
    for (int j = 0; j < 4; ++j) acc[i][j] = wmma_bf16(a0[i], b0[j], acc[i][j]);
#pragma unroll
  for (int i = 0; i < 4; ++i)
#pragma unroll
    for (int j = 0; j < 4; ++j) acc[i][j] = wmma_bf16(a1[i], b1[j], acc[i][j]);

#pragma unroll
  for (int i = 0; i < 4; ++i) {
    const int mb = row0 + wm * 64 + i * 16 + (hi ? 8 : 0);
#pragma unroll
    for (int j = 0; j < 4; ++j) {
      const int n = n0 + j * 16 + l15;
      const float bias = be1[e * FH + n];
#pragma unroll
      for (int r = 0; r < 8; ++r) {
        const float c = acc[i][j][r] + bias;
        const float s = c / (1.f + expf(-c));     // silu
        Hbuf[(size_t)(mb + r) * FH + n] = f2bf(s);
      }
    }
  }
}

// ---------------------------------------------------------------- MoE GEMM2
__global__ __launch_bounds__(256) void gemm2_moe_kernel(
    const unsigned short* __restrict__ Hbuf,  // [CAP][FH]
    const unsigned short* __restrict__ We2T,  // [NE][DM][FH]
    const float* __restrict__ be2,            // [NE][DM]
    const int* __restrict__ base_pad,
    float* __restrict__ Obuf) {               // [CAP][DM]
  const int row0 = blockIdx.x * 128;
  if (row0 >= base_pad[NE]) return;
  int e = 0;
#pragma unroll
  for (int i = 0; i < NE; ++i) if (row0 >= base_pad[i + 1]) e = i + 1;
  const int tid = threadIdx.x, wid = tid >> 5, lane = tid & 31;
  const int wm = wid >> 2, wn = wid & 3;
  const int l15 = lane & 15, hi = lane >> 4;
  const unsigned short* ap[4];
#pragma unroll
  for (int i = 0; i < 4; ++i)
    ap[i] = Hbuf + (size_t)(row0 + wm * 64 + i * 16 + l15) * FH + hi * 16;
  const int n0 = blockIdx.y * 256 + wn * 64;
  const unsigned short* bbase = We2T + (size_t)e * DM * FH;
  const unsigned short* bp[4];
#pragma unroll
  for (int j = 0; j < 4; ++j) bp[j] = bbase + (size_t)(n0 + j * 16 + l15) * FH + hi * 16;
  v8f acc[4][4];
#pragma unroll
  for (int i = 0; i < 4; ++i)
#pragma unroll
    for (int j = 0; j < 4; ++j)
#pragma unroll
      for (int r = 0; r < 8; ++r) acc[i][j][r] = 0.f;

  v16bf a0[4], a1[4], b0[4], b1[4];
#pragma unroll
  for (int i = 0; i < 4; ++i) a0[i] = load_a(ap[i]);
#pragma unroll
  for (int j = 0; j < 4; ++j) b0[j] = load_b(bp[j]);
  for (int kk = 0; kk < FH - 64; kk += 64) {
#pragma unroll
    for (int i = 0; i < 4; ++i) a1[i] = load_a(ap[i] + kk + 32);
#pragma unroll
    for (int j = 0; j < 4; ++j) b1[j] = load_b(bp[j] + kk + 32);
#pragma unroll
    for (int j = 0; j < 4; ++j) __builtin_prefetch(bp[j] + kk + 128, 0, 1);
#pragma unroll
    for (int i = 0; i < 4; ++i)
#pragma unroll
      for (int j = 0; j < 4; ++j) acc[i][j] = wmma_bf16(a0[i], b0[j], acc[i][j]);
#pragma unroll
    for (int i = 0; i < 4; ++i) a0[i] = load_a(ap[i] + kk + 64);
#pragma unroll
    for (int j = 0; j < 4; ++j) b0[j] = load_b(bp[j] + kk + 64);
#pragma unroll
    for (int i = 0; i < 4; ++i)
#pragma unroll
      for (int j = 0; j < 4; ++j) acc[i][j] = wmma_bf16(a1[i], b1[j], acc[i][j]);
  }
#pragma unroll
  for (int i = 0; i < 4; ++i) a1[i] = load_a(ap[i] + FH - 32);
#pragma unroll
  for (int j = 0; j < 4; ++j) b1[j] = load_b(bp[j] + FH - 32);
#pragma unroll
  for (int i = 0; i < 4; ++i)
#pragma unroll
    for (int j = 0; j < 4; ++j) acc[i][j] = wmma_bf16(a0[i], b0[j], acc[i][j]);
#pragma unroll
  for (int i = 0; i < 4; ++i)
#pragma unroll
    for (int j = 0; j < 4; ++j) acc[i][j] = wmma_bf16(a1[i], b1[j], acc[i][j]);

#pragma unroll
  for (int i = 0; i < 4; ++i) {
    const int mb = row0 + wm * 64 + i * 16 + (hi ? 8 : 0);
#pragma unroll
    for (int j = 0; j < 4; ++j) {
      const int n = n0 + j * 16 + l15;
      const float bias = be2[e * DM + n];
#pragma unroll
      for (int r = 0; r < 8; ++r)
        Obuf[(size_t)(mb + r) * DM + n] = acc[i][j][r] + bias;
    }
  }
}

__global__ __launch_bounds__(256) void moe_combine_kernel(
    const float* __restrict__ Obuf, const int* __restrict__ slot_of,
    const float* __restrict__ tok_w, float* __restrict__ xout) {
  const int t = blockIdx.x, tid = threadIdx.x;
  const int s0 = slot_of[2 * t], s1 = slot_of[2 * t + 1];
  const float w0 = tok_w[2 * t], w1 = tok_w[2 * t + 1];
  const float4 o0 = ((const float4*)(Obuf + (size_t)s0 * DM))[tid];
  const float4 o1 = ((const float4*)(Obuf + (size_t)s1 * DM))[tid];
  float4 xv = ((float4*)(xout + (size_t)t * DM))[tid];
  xv.x += w0 * o0.x + w1 * o1.x;
  xv.y += w0 * o0.y + w1 * o1.y;
  xv.z += w0 * o0.z + w1 * o1.z;
  xv.w += w0 * o0.w + w1 * o1.w;
  ((float4*)(xout + (size_t)t * DM))[tid] = xv;
}

// ================================================================ launch
extern "C" void kernel_launch(void* const* d_in, const int* in_sizes, int n_in,
                              void* d_out, int out_size, void* d_ws, size_t ws_size,
                              hipStream_t stream) {
  (void)in_sizes; (void)n_in; (void)out_size; (void)ws_size;
  const float* x        = (const float*)d_in[0];
  const float* state0   = (const float*)d_in[1];
  const float* tau1     = (const float*)d_in[2];
  const float* scale1   = (const float*)d_in[3];
  const float* tau2     = (const float*)d_in[4];
  const float* scale2   = (const float*)d_in[5];
  const float* W_in     = (const float*)d_in[6];
  const float* b_in     = (const float*)d_in[7];
  const float* rec_w    = (const float*)d_in[8];
  const float* pos_bias = (const float*)d_in[9];
  const float* W_out    = (const float*)d_in[10];
  const float* b_out    = (const float*)d_in[11];
  const float* W_router = (const float*)d_in[12];
  const float* We1      = (const float*)d_in[13];
  const float* be1      = (const float*)d_in[14];
  const float* We2      = (const float*)d_in[15];
  const float* be2      = (const float*)d_in[16];

  float* xout      = (float*)d_out;                 // [NTOK][DM]
  float* out_state = xout + (size_t)NTOK * DM;      // [NB][DS]
  float* out_aux   = out_state + NB * DS;           // [1]

  char* base = (char*)d_ws; size_t off = 0;
  auto alloc = [&](size_t nbytes) -> void* {
    off = (off + 255) & ~(size_t)255;
    void* p = base + off; off += nbytes; return p;
  };
  unsigned short* WinT   = (unsigned short*)alloc((size_t)128 * DM * 2);
  unsigned short* WoutT  = (unsigned short*)alloc((size_t)DM * DS * 2);
  unsigned short* We1T   = (unsigned short*)alloc((size_t)NE * FH * DM * 2);
  unsigned short* We2T   = (unsigned short*)alloc((size_t)NE * DM * FH * 2);
  unsigned short* hbf    = (unsigned short*)alloc((size_t)NTOK * DM * 2);  // h1 then h2
  float* u_buf           = (float*)alloc((size_t)NTOK * DS * 4);
  float* states          = (float*)alloc((size_t)NTOK * DS * 4);
  unsigned short* sbf    = (unsigned short*)alloc((size_t)NTOK * DS * 2);
  float* chunk_last      = (float*)alloc((size_t)NB * NCH * DS * 4);
  float* chunk_start     = (float*)alloc((size_t)NB * NCH * DS * 4);
  int*   counts          = (int*)alloc(NE * 4);
  int*   counts2         = (int*)alloc(NE * 4);
  int*   base_pad        = (int*)alloc((NE + 1) * 4);
  float* Ppart           = (float*)alloc((size_t)1024 * NE * 4);
  int*   tok_e           = (int*)alloc((size_t)NTOK * 2 * 4);
  float* tok_w           = (float*)alloc((size_t)NTOK * 2 * 4);
  int*   slot_of         = (int*)alloc((size_t)NTOK * 2 * 4);
  int*   assign_tok      = (int*)alloc((size_t)CAP * 4);
  unsigned short* Hbuf   = (unsigned short*)alloc((size_t)CAP * FH * 2);
  float* Obuf            = (float*)alloc((size_t)CAP * DM * 4);

  const dim3 blk(256);
  // weight transposes -> bf16 N-major
  transpose_bf16_kernel<<<dim3(128 / 32, DM / 32, 1), blk, 0, stream>>>(
      W_in, WinT, DM, 128, 0, 0);
  transpose_bf16_kernel<<<dim3(DM / 32, DS / 32, 1), blk, 0, stream>>>(
      W_out, WoutT, DS, DM, 0, 0);
  transpose_bf16_kernel<<<dim3(FH / 32, DM / 32, NE), blk, 0, stream>>>(
      We1, We1T, DM, FH, (size_t)DM * FH, (size_t)FH * DM);
  transpose_bf16_kernel<<<dim3(DM / 32, FH / 32, NE), blk, 0, stream>>>(
      We2, We2T, FH, DM, (size_t)FH * DM, (size_t)DM * FH);
  init_counts_kernel<<<1, 32, 0, stream>>>(counts);

  // compnorm1 -> h1 (bf16)
  compnorm_kernel<<<NTOK, blk, 0, stream>>>(x, tau1, scale1, hbf);
  // proj GEMM + gates -> u
  gemm_proj_kernel<<<NTOK / 128, blk, 0, stream>>>(hbf, WinT, b_in, u_buf);
  // LRU scan
  lru_chunk_kernel<<<(NB * NCH * DS) / 256, blk, 0, stream>>>(
      u_buf, rec_w, pos_bias, states, chunk_last);
  lru_combine_kernel<<<1, blk, 0, stream>>>(
      rec_w, pos_bias, state0, chunk_last, chunk_start, out_state);
  lru_fixup_kernel<<<(NB * TT * DS) / 256, blk, 0, stream>>>(
      states, chunk_start, rec_w, pos_bias, sbf);
  // out GEMM + residual -> x1 in d_out
  gemm_out_kernel<<<dim3(NTOK / 128, DM / 128), blk, 0, stream>>>(
      sbf, WoutT, b_out, x, xout);
  // compnorm2 -> h2 (bf16, reuse buffer)
  compnorm_kernel<<<NTOK, blk, 0, stream>>>(xout, tau2, scale2, hbf);
  // router
  router_kernel<<<NTOK / 8, blk, 0, stream>>>(hbf, W_router, counts, Ppart, tok_e, tok_w);
  router_setup_kernel<<<1, blk, 0, stream>>>(counts, Ppart, base_pad, counts2,
                                             assign_tok, out_aux);
  router_scatter_kernel<<<NTOK / 256, blk, 0, stream>>>(tok_e, counts2, base_pad,
                                                        assign_tok, slot_of);
  // expert FFN: block 128M x 256N, wave 64M x 64N
  gemm1_moe_kernel<<<dim3(CAP / 128, FH / 256), blk, 0, stream>>>(
      hbf, We1T, be1, assign_tok, base_pad, Hbuf);
  gemm2_moe_kernel<<<dim3(CAP / 128, DM / 256), blk, 0, stream>>>(
      Hbuf, We2T, be2, base_pad, Obuf);
  moe_combine_kernel<<<NTOK, blk, 0, stream>>>(Obuf, slot_of, tok_w, xout);
}
